// MapNet_6468220748262
// MI455X (gfx1250) — matrix operand order repaired and checked by reading, not compiled, not run on hardware
//
#include <hip/hip_runtime.h>

typedef __attribute__((ext_vector_type(16))) _Float16 v16h;
typedef __attribute__((ext_vector_type(8)))  float    v8f;

#define HIDDEN  256
#define KOUT    27
#define P_TOT   (1024*1024)
#define NCHUNK  (P_TOT/16)
#define TPB     256
#define NBLK    1024

// feats is (3,512,512) row-major; zero-pad outside [0,512)^2 (matches jnp.pad border).
__device__ __forceinline__ float fetch_feat(const float* __restrict__ f, int c, int y, int x) {
  if ((unsigned)y < 512u && (unsigned)x < 512u)
    return f[(c << 18) + (y << 9) + x];
  return 0.f;
}

// exact leaky_relu(0.01): x>=0 -> x ; x<0 -> 0.01x   ==  0.505x + 0.495|x|
__device__ __forceinline__ float leaky(float x) {
  return fmaf(fabsf(x), 0.495f, x * 0.505f);
}

__global__ __launch_bounds__(TPB) void mapnet_fused(
    const float* __restrict__ pos,      // P x 3
    const long long* __restrict__ mp,   // P x 2 (int64 y,x)
    const float* __restrict__ feats,    // 3 x 512 x 512
    const float* __restrict__ W1,       // 256 x 3
    const float* __restrict__ b1,       // 256
    const float* __restrict__ W2,       // 27 x 256
    const float* __restrict__ b2,       // 27
    float* __restrict__ out)            // P
{
  // Pre-swizzled constant fragments, filled once per block.
  // sA1[mt][lane][i]: stage-1 A frag (W1 rows = hidden, K=0..2 = pos dims, K=3 = b1, rest 0)
  // sB2[2t+nt][lane][i]: stage-2 B frag (W2T: K = hidden, N = out-k, N>=27 zero-padded)
  __shared__ __align__(32) _Float16 sA1[16][32][16];   // 16 KB
  __shared__ __align__(32) _Float16 sB2[16][32][16];   // 16 KB

  const int tid = threadIdx.x;
  // Stage W1(+b1) A-fragments: one whole v16h per (mt,lane) -> 2x ds_store_b128.
  for (int fi = tid; fi < 16*32; fi += TPB) {
    int ln = fi & 31, mt = fi >> 5;
    int kb = (ln < 16) ? 0 : 8;               // A-frag K-base per half-wave
    int h  = mt * 16 + (ln & 15);             // A row M = lane&15
    v16h frag;
    #pragma unroll
    for (int i = 0; i < 16; ++i) {
      int K = kb + ((i >> 3) << 4) + (i & 7); // K = kb + 16*(i/8) + i%8
      float v = 0.f;
      if (K < 3)       v = W1[h*3 + K];
      else if (K == 3) v = b1[h];             // bias folded into K=3 (B supplies 1.0)
      frag[i] = (_Float16)v;
    }
    *(v16h*)&sA1[mt][ln][0] = frag;
  }
  // Stage W2 B-fragments.
  for (int fi = tid; fi < 16*32; fi += TPB) {
    int ln = fi & 31, f = fi >> 5;
    int t = f >> 1, nt = f & 1;
    int n  = nt*16 + (ln & 15);               // B col N = lane&15
    int Kb = t*32 + ((ln < 16) ? 0 : 8);      // K = Kb + 16*(i/8) + i%8
    v16h frag;
    #pragma unroll
    for (int i = 0; i < 16; ++i) {
      float v = (n < KOUT) ? W2[n*HIDDEN + Kb + ((i >> 3) << 4) + (i & 7)] : 0.f;
      frag[i] = (_Float16)v;
    }
    *(v16h*)&sB2[f][ln][0] = frag;
  }
  __syncthreads();

  const int lane  = tid & 31;
  const int hh    = lane >> 4;     // half-wave id (C-layout: M = r + 8*hh)
  const int ncol  = lane & 15;     // C-layout N; also the pixel this lane feeds in stage-1 B
  const int wave  = blockIdx.x * (TPB/32) + (tid >> 5);
  const int wstr  = NBLK * (TPB/32);

  // per-lane constants
  const float b20 = b2[ncol];
  const float b21 = (ncol + 16 < KOUT) ? b2[ncol + 16] : 0.f;
  // stage-3: this lane owns lw columns k0 = ncol, k1 = ncol+16
  const int  k0  = ncol;
  const int  c0  = k0 / 9, q0 = k0 % 9, di0 = q0/3 - 1, dj0 = q0%3 - 1;
  const bool k1v = (ncol + 16) < KOUT;
  const int  k1  = k1v ? (ncol + 16) : 0;
  const int  c1  = k1 / 9, q1 = k1 % 9, di1 = q1/3 - 1, dj1 = q1%3 - 1;
  const float lmask = (hh == 0) ? 1.f : 0.f;   // stage-1 B lives in K=0..3 -> lanes 0..15 only

  unsigned zr = 0;   // opaque zero: defeats LICM of the LDS fragment loads (keeps VGPRs < 256)

  for (int chunk = wave; chunk < NCHUNK; chunk += wstr) {
    asm volatile("" : "+s"(zr));   // zr is still 0, but the compiler must reload frags each iter
    const int p0 = chunk * 16;

    // prefetch next chunk's per-pixel inputs into near caches (global_prefetch_b8)
    const int nxt = chunk + wstr;
    if (nxt < NCHUNK) {
      __builtin_prefetch(&pos[(size_t)(nxt*16 + ncol) * 3], 0, 3);
      __builtin_prefetch(&mp[(size_t)(nxt*16 + ncol) * 2], 0, 3);
    }

    // ---- stage-1 B fragment: posT (K=0..2) + ones row (K=3 -> +b1) ----
    const int pp = p0 + ncol;
    const float px = pos[(size_t)pp*3 + 0];
    const float py = pos[(size_t)pp*3 + 1];
    const float pz = pos[(size_t)pp*3 + 2];
    v16h bpos;
    #pragma unroll
    for (int i = 0; i < 16; ++i) bpos[i] = (_Float16)0.f;
    bpos[0] = (_Float16)(px * lmask);
    bpos[1] = (_Float16)(py * lmask);
    bpos[2] = (_Float16)(pz * lmask);
    bpos[3] = (_Float16)lmask;

    // stage-2 accumulators: lw[m = r+8*hh][n = ncol (+16)]
    v8f c2a, c2b;
    #pragma unroll
    for (int i = 0; i < 8; ++i) { c2a[i] = b20; c2b[i] = b21; }

    // ---- fused MLP: per K-tile t, two stage-1 WMMAs produce hT tiles whose
    // C-register layout IS the stage-2 A fragment after leaky + f32->f16 ----
    #pragma unroll
    for (int t = 0; t < 8; ++t) {
      v16h w1a = *(const v16h*)&sA1[2*t     + zr][lane][0];
      v16h w1b = *(const v16h*)&sA1[2*t + 1 + zr][lane][0];
      v8f z = {};
      v8f h0 = __builtin_amdgcn_wmma_f32_16x16x32_f16(false, w1a, false, bpos,
                                                      (short)0, z, false, false);
      v8f h1 = __builtin_amdgcn_wmma_f32_16x16x32_f16(false, w1b, false, bpos,
                                                      (short)0, z, false, false);
      v16h af;
      #pragma unroll
      for (int i = 0; i < 8; ++i) {
        af[i]     = (_Float16)leaky(h0[i]);   // K = 32t + 8*hh + i
        af[i + 8] = (_Float16)leaky(h1[i]);   // K = 32t + 8*hh + 16 + i
      }
      v16h wb0 = *(const v16h*)&sB2[2*t     + zr][lane][0];
      v16h wb1 = *(const v16h*)&sB2[2*t + 1 + zr][lane][0];
      c2a = __builtin_amdgcn_wmma_f32_16x16x32_f16(false, af, false, wb0,
                                                   (short)0, c2a, false, false);
      c2b = __builtin_amdgcn_wmma_f32_16x16x32_f16(false, af, false, wb1,
                                                   (short)0, c2b, false, false);
    }

    // ---- stage-3: gather feats taps, dot with lw, reduce across 16 lanes ----
    float acc[8];
    #pragma unroll
    for (int r = 0; r < 8; ++r) {
      const int m = r + 8*hh;
      const size_t p = (size_t)(p0 + m);
      const int y = (int)mp[p*2 + 0];
      const int x = (int)mp[p*2 + 1];
      const float f0 = fetch_feat(feats, c0, y + di0, x + dj0);
      const float f1 = k1v ? fetch_feat(feats, c1, y + di1, x + dj1) : 0.f;
      acc[r] = f0 * c2a[r] + f1 * c2b[r];
    }
    #pragma unroll
    for (int off = 1; off < 16; off <<= 1) {
      #pragma unroll
      for (int r = 0; r < 8; ++r)
        acc[r] += __shfl_xor(acc[r], off, 32);
    }
    if (ncol == 0) {   // lanes 0 and 16 each hold 8 contiguous pixel sums
      float* dst = out + p0 + 8*hh;
      float4 o0 = make_float4(acc[0], acc[1], acc[2], acc[3]);
      float4 o1 = make_float4(acc[4], acc[5], acc[6], acc[7]);
      *(float4*)(dst)     = o0;
      *(float4*)(dst + 4) = o1;
    }
  }
}

extern "C" void kernel_launch(void* const* d_in, const int* in_sizes, int n_in,
                              void* d_out, int out_size, void* d_ws, size_t ws_size,
                              hipStream_t stream) {
  const float*     pos   = (const float*)d_in[0];
  const long long* mpmat = (const long long*)d_in[1];
  const float*     feats = (const float*)d_in[2];
  // d_in[3] = depth: unused by the reference computation
  const float*     W1    = (const float*)d_in[4];
  const float*     b1    = (const float*)d_in[5];
  const float*     W2    = (const float*)d_in[6];
  const float*     b2    = (const float*)d_in[7];
  mapnet_fused<<<NBLK, TPB, 0, stream>>>(pos, mpmat, feats, W1, b1, W2, b2,
                                         (float*)d_out);
}
